// SwinTransformer_15693810500359
// MI455X (gfx1250) — matrix operand order repaired and checked by reading, not compiled, not run on hardware
//
#include <hip/hip_runtime.h>
#include <hip/hip_bf16.h>

// ---------------------------------------------------------------------------
// Swin-style attention block for MI455X (gfx1250, wave32, WMMA).
// B=8, N=1024, D=1024, H=8, hd=128.  All GEMMs run on v_wmma_f32_16x16x32_bf16.
// X tiles double-buffered via global_load_async_to_lds_b128 (ASYNCcnt) so the
// LDS DMA overlaps the WMMAs; V stored transposed so every WMMA operand is
// fed by b128 loads.
// ---------------------------------------------------------------------------

#define BB 8
#define NN 1024
#define DD 1024
#define HH 8
#define HD 128
#define BN (BB * NN)       // 8192 rows of X
#define BH (BB * HH)       // 64 (batch, head) pairs

typedef __attribute__((ext_vector_type(16))) __bf16 v16bf;
typedef __attribute__((ext_vector_type(8)))  float  v8f;
typedef __attribute__((ext_vector_type(8)))  unsigned short us8;

union AFrag {
    us8            h2[2];   // two 16-byte halves
    unsigned short u[16];
    v16bf          v;
};

static __device__ __forceinline__ unsigned short f32_to_bf16(float f) {
    unsigned int x = __builtin_bit_cast(unsigned int, f);
    unsigned int r = x + 0x7FFFu + ((x >> 16) & 1u);   // round-to-nearest-even
    return (unsigned short)(r >> 16);
}

static __device__ __forceinline__ float bf16_to_f32(unsigned short u) {
    return __builtin_bit_cast(float, ((unsigned int)u) << 16);
}

static __device__ __forceinline__ v8f wmma_bf16(v16bf a, v16bf b, v8f c) {
    // D = A(16x32 bf16) x B(32x16 bf16) + C(16x16 f32)
    return __builtin_amdgcn_wmma_f32_16x16x32_bf16(
        /*neg_a=*/false, a, /*neg_b=*/false, b,
        /*c_mod=*/(short)0, c, /*reuse_a=*/false, /*reuse_b=*/false);
}

static __device__ __forceinline__ void async_tile_b128(unsigned int lds_addr,
                                                       const unsigned short* g) {
    asm volatile("global_load_async_to_lds_b128 %0, %1, off"
                 :: "v"(lds_addr), "v"(g) : "memory");
}

static __device__ __forceinline__ void wait_asynccnt0() {
    asm volatile("s_wait_asynccnt 0x0" ::: "memory");
}

// ---------------------------------------------------------------------------
// Kernel 1: fp32 -> bf16 conversion of X (windows) and Wq/Wk/Wv.
// ---------------------------------------------------------------------------
__global__ void convert_bf16_kernel(const float* __restrict__ X,
                                    const float* __restrict__ Wq,
                                    const float* __restrict__ Wk,
                                    const float* __restrict__ Wv,
                                    unsigned short* __restrict__ Xb,
                                    unsigned short* __restrict__ Wb) {
    const long long NX = (long long)BN * DD;   // 8,388,608
    const long long NW = (long long)DD * DD;   // 1,048,576
    const long long total = NX + 3 * NW;
    long long i = (long long)blockIdx.x * blockDim.x + threadIdx.x;
    const long long stride = (long long)gridDim.x * blockDim.x;
    for (; i < total; i += stride) {
        if (i < NX) {
            Xb[i] = f32_to_bf16(X[i]);
        } else {
            long long j = i - NX;
            const float* W = (j < NW) ? Wq : (j < 2 * NW ? Wk : Wv);
            Wb[j] = f32_to_bf16(W[j % NW]);
        }
    }
}

// ---------------------------------------------------------------------------
// Kernel 2: QKV projection GEMM.  Y = X * W^T + bias.
// Block = 256 threads (8 waves).  Block tile = 64 rows x 128 cols; each wave
// owns FOUR 16x16 output tiles, so one staged X[64x32] LDS tile and one B
// fragment feed 4 WMMAs per barrier.  X tiles are DOUBLE-BUFFERED through
// global_load_async_to_lds_b128: the k0+32 DMA is issued right after the
// barrier and lands while the current tile's WMMAs execute; s_wait_asynccnt
// only gates the *next* iteration.  One barrier per K-step.
// B[k][n] = W[j0+n][k] is contiguous in k -> two b128 loads, no transpose.
// Q/K written bf16 as [B,H,N,HD] (scale 1/32 folded into Q);
// V written bf16 TRANSPOSED as [B,H,HD,N] (vector b128 stores).
// grid = (BN/64, DD/128, 3), blockIdx.z selects q/k/v.
// ---------------------------------------------------------------------------
__global__ __launch_bounds__(256) void qkv_gemm_kernel(
        const unsigned short* __restrict__ Xb,
        const unsigned short* __restrict__ Wb,
        const float* __restrict__ bq, const float* __restrict__ bk,
        const float* __restrict__ bv,
        unsigned short* __restrict__ Qb, unsigned short* __restrict__ Kb,
        unsigned short* __restrict__ Vt) {
    const int mat = blockIdx.z;  // 0=q 1=k 2=v
    const unsigned short* W = Wb + (size_t)mat * DD * DD;
    const float* bias = (mat == 0) ? bq : (mat == 1) ? bk : bv;
    unsigned short* Out = (mat == 0) ? Qb : (mat == 1) ? Kb : Vt;

    const int rows0 = blockIdx.x * 64;
    const int wave  = threadIdx.x >> 5;
    const int lane  = threadIdx.x & 31;
    const int j0    = blockIdx.y * 128 + wave * 16;

    __shared__ unsigned short Xs[2][64 * 32];   // 2 x 4 KB ping-pong tiles

    const int m   = lane & 15;          // A: matrix row per lane
    const int kbA = (lane >> 4) * 8;    // A: K half select
    const int nl  = lane & 15;          // B/C: matrix col per lane
    const int kbB = (lane >> 4) * 8;    // B: K half select

    v8f accs[4];
    #pragma unroll
    for (int rt = 0; rt < 4; ++rt) accs[rt] = (v8f)0.0f;

    // Cooperative staging assignment: 256 threads x 16B = 64x32 bf16 tile.
    const int lr = threadIdx.x >> 2;          // 0..63
    const int lc = (threadIdx.x & 3) * 8;     // 0,8,16,24
    const unsigned short* gX = &Xb[(size_t)(rows0 + lr) * DD + lc];
    unsigned int laddr[2];
    laddr[0] = (unsigned int)(unsigned long long)(void*)&Xs[0][lr * 32 + lc];
    laddr[1] = (unsigned int)(unsigned long long)(void*)&Xs[1][lr * 32 + lc];

    // Prologue: kick off the first tile's DMA.
    async_tile_b128(laddr[0], gX);

    for (int k0 = 0; k0 < DD; k0 += 32) {
        const int cur = (k0 >> 5) & 1;
        wait_asynccnt0();        // my slice of Xs[cur] has landed
        __syncthreads();         // all slices visible; Xs[1-cur] free to reuse
        if (k0 + 32 < DD)        // overlap next DMA with this tile's WMMAs
            async_tile_b128(laddr[1 - cur], gX + k0 + 32);

        AFrag b;                               // B = W^T tile (global, L2-hot)
        const unsigned short* wrow = &W[(size_t)(j0 + nl) * DD + k0 + kbB];
        __builtin_prefetch(wrow + 128, 0, 1);  // next K panel of this W row
        b.h2[0] = *(const us8*)&wrow[0];
        b.h2[1] = *(const us8*)&wrow[16];

        const unsigned short* xs = Xs[cur];
        #pragma unroll
        for (int rt = 0; rt < 4; ++rt) {       // 4 WMMAs reuse one B fragment
            AFrag a;
            a.h2[0] = *(const us8*)&xs[(rt * 16 + m) * 32 + kbA];
            a.h2[1] = *(const us8*)&xs[(rt * 16 + m) * 32 + kbA + 16];
            accs[rt] = wmma_bf16(a.v, b.v, accs[rt]);
        }
    }

    const float bval  = bias[j0 + nl];
    const int   mbase = (lane >> 4) * 8;
    const int   j     = j0 + nl;
    const int   hh    = j >> 7;       // head
    const int   t     = j & 127;      // within-head col
    if (mat < 2) {
        const float sc = (mat == 0) ? 0.03125f : 1.0f;  // 1/sqrt(D) into Q
        #pragma unroll
        for (int rt = 0; rt < 4; ++rt) {
            #pragma unroll
            for (int v = 0; v < 8; ++v) {
                const int r  = rows0 + rt * 16 + mbase + v;
                const int bi = r >> 10;
                const int n  = r & 1023;
                Out[(((size_t)(bi * HH + hh)) * NN + n) * HD + t] =
                    f32_to_bf16((accs[rt][v] + bval) * sc);
            }
        }
    } else {
        // V transposed: Vt[((b*H+h)*HD + t)*N + n]; 8 consecutive n per lane
        // -> one 16-byte store per row tile.
        #pragma unroll
        for (int rt = 0; rt < 4; ++rt) {
            const int rbase = rows0 + rt * 16 + mbase;   // multiple of 8
            const int bi = rbase >> 10;
            const int nb = rbase & 1023;
            us8 pack;
            #pragma unroll
            for (int v = 0; v < 8; ++v)
                pack[v] = f32_to_bf16(accs[rt][v] + bval);
            *(us8*)&Out[(((size_t)(bi * HH + hh)) * HD + t) * NN + nb] = pack;
        }
    }
}

// ---------------------------------------------------------------------------
// Kernel 3: fused attention for one (b,h) and one 16-query tile.
// Phase 1: waves split the 64 key tiles of Q*K^T (4 bf16 WMMAs over hd=128),
// logits + position_bias -> bf16 LDS tile P[16][1024] (32 KB).
// Phase 2: row softmax (16 lanes/row, shfl_xor reduce); diagonal overwritten
// with bf16 -inf (post-softmax mask, faithful to ref); 1/Z per row.
// Phase 3: P*V via WMMA with V read from the transposed layout (b128 loads),
// scaled by 1/Z on f32 output.  grid = (N/16, B*H).
// ---------------------------------------------------------------------------
__global__ __launch_bounds__(256) void attn_kernel(
        const unsigned short* __restrict__ Qb,
        const unsigned short* __restrict__ Kb,
        const unsigned short* __restrict__ Vt,
        const float* __restrict__ pos_bias,
        float* __restrict__ out) {
    const int n0   = blockIdx.x * 16;
    const int bh   = blockIdx.y;
    const int b    = bh >> 3;
    const int h    = bh & 7;
    const int wave = threadIdx.x >> 5;
    const int lane = threadIdx.x & 31;

    __shared__ unsigned short P[16][1024];   // 32 KB: bf16 probabilities
    __shared__ float rowInv[16];             // per-row 1/Z

    const unsigned short* Qh  = Qb + (size_t)bh * NN * HD;
    const unsigned short* Kh  = Kb + (size_t)bh * NN * HD;
    const unsigned short* Vth = Vt + (size_t)bh * HD * NN;   // [HD][N]

    const int mrow  = lane & 15;
    const int kbA   = (lane >> 4) * 8;
    const int nl    = lane & 15;
    const int kbB   = (lane >> 4) * 8;
    const int mbase = (lane >> 4) * 8;

    // Preload the 16x128 Q tile as four 16x32 A-fragments (shared by phase 1).
    AFrag qf[4];
    #pragma unroll
    for (int kc = 0; kc < 4; ++kc) {
        const unsigned short* qrow =
            &Qh[(size_t)(n0 + mrow) * HD + kc * 32 + kbA];
        qf[kc].h2[0] = *(const us8*)&qrow[0];
        qf[kc].h2[1] = *(const us8*)&qrow[16];
    }

    // ---- Phase 1: logits S = Q*K^T (scale already folded into Q) ----------
    for (int mt = 0; mt < 8; ++mt) {
        const int m0 = (mt * 8 + wave) * 16;
        __builtin_prefetch(&Kh[(size_t)(m0 + 128 + nl) * HD], 0, 1);
        v8f acc = (v8f)0.0f;
        #pragma unroll
        for (int kc = 0; kc < 4; ++kc) {
            AFrag bkf;   // B[k=d][n=m] = K[m0+n][d]: contiguous in d
            const unsigned short* krow =
                &Kh[(size_t)(m0 + nl) * HD + kc * 32 + kbB];
            bkf.h2[0] = *(const us8*)&krow[0];
            bkf.h2[1] = *(const us8*)&krow[16];
            acc = wmma_bf16(qf[kc].v, bkf.v, acc);
        }
        #pragma unroll
        for (int v = 0; v < 8; ++v) {
            const float pb =
                pos_bias[(size_t)(n0 + mbase + v) * DD + (m0 + nl)];
            P[mbase + v][m0 + nl] = f32_to_bf16(acc[v] + pb);
        }
    }
    __syncthreads();

    // ---- Phase 2: row softmax + post-softmax diagonal mask -----------------
    {
        const int row = threadIdx.x >> 4;   // 0..15 query row
        const int l16 = threadIdx.x & 15;   // 16 lanes cooperate per row
        float mx = -3.4e38f;
        for (int c = l16; c < NN; c += 16)
            mx = fmaxf(mx, bf16_to_f32(P[row][c]));
        for (int off = 8; off; off >>= 1)
            mx = fmaxf(mx, __shfl_xor(mx, off, 32));   // masks<16 stay in-row
        float sum = 0.f;
        for (int c = l16; c < NN; c += 16)
            sum += __expf(bf16_to_f32(P[row][c]) - mx);
        for (int off = 8; off; off >>= 1)
            sum += __shfl_xor(sum, off, 32);
        if (l16 == 0) rowInv[row] = 1.0f / sum;
        for (int c = l16; c < NN; c += 16) {
            unsigned short p =
                f32_to_bf16(__expf(bf16_to_f32(P[row][c]) - mx));
            if (c == n0 + row) p = 0xFF80u;  // bf16 -inf: masked_fill after softmax
            P[row][c] = p;
        }
    }
    __syncthreads();

    // ---- Phase 3: O = (P * V) * (1/Z); each wave owns 16 head-dim cols -----
    const int t0 = wave * 16;
    v8f acc = (v8f)0.0f;
    for (int m0 = 0; m0 < NN; m0 += 32) {
        AFrag a;    // A = P tile from LDS (ds_load_b128 x2)
        a.h2[0] = *(const us8*)&P[mrow][m0 + kbA];
        a.h2[1] = *(const us8*)&P[mrow][m0 + kbA + 16];
        AFrag bvf;  // B[k=m][n=t] = Vt[t0+n][m0+k]: contiguous in k
        const unsigned short* vrow = &Vth[(size_t)(t0 + nl) * NN + m0 + kbB];
        bvf.h2[0] = *(const us8*)&vrow[0];
        bvf.h2[1] = *(const us8*)&vrow[16];
        acc = wmma_bf16(a.v, bvf.v, acc);
    }
    const int col = h * HD + t0 + nl;
    #pragma unroll
    for (int v = 0; v < 8; ++v) {
        const int nrow = n0 + mbase + v;
        out[((size_t)b * NN + nrow) * DD + col] = acc[v] * rowInv[mbase + v];
    }
}

// ---------------------------------------------------------------------------
// Launch
// ---------------------------------------------------------------------------
extern "C" void kernel_launch(void* const* d_in, const int* in_sizes, int n_in,
                              void* d_out, int out_size, void* d_ws, size_t ws_size,
                              hipStream_t stream) {
    const float* windows  = (const float*)d_in[0];
    const float* Wq       = (const float*)d_in[1];
    const float* bq       = (const float*)d_in[2];
    const float* Wk       = (const float*)d_in[3];
    const float* bk       = (const float*)d_in[4];
    const float* Wv       = (const float*)d_in[5];
    const float* bv       = (const float*)d_in[6];
    const float* pos_bias = (const float*)d_in[7];
    // d_in[8] = mask (eye(D)); the diagonal mask is applied structurally.
    float* out = (float*)d_out;

    // Workspace layout (bytes): Xb 16MB | Wb 6MB | Qb 16MB | Kb 16MB | Vt 16MB
    char* ws = (char*)d_ws;
    const size_t szX = (size_t)BN * DD * sizeof(unsigned short);
    const size_t szW = 3 * (size_t)DD * DD * sizeof(unsigned short);
    const size_t szH = (size_t)BH * NN * HD * sizeof(unsigned short);
    unsigned short* Xb = (unsigned short*)(ws);
    unsigned short* Wb = (unsigned short*)(ws + szX);
    unsigned short* Qb = (unsigned short*)(ws + szX + szW);
    unsigned short* Kb = (unsigned short*)(ws + szX + szW + szH);
    unsigned short* Vt = (unsigned short*)(ws + szX + szW + 2 * szH);

    convert_bf16_kernel<<<8192, 256, 0, stream>>>(windows, Wq, Wk, Wv, Xb, Wb);

    dim3 g2(BN / 64, DD / 128, 3);
    qkv_gemm_kernel<<<g2, 256, 0, stream>>>(Xb, Wb, bq, bk, bv, Qb, Kb, Vt);

    dim3 g3(NN / 16, BH);
    attn_kernel<<<g3, 256, 0, stream>>>(Qb, Kb, Vt, pos_bias, out);
}
